// WAM_13056700579937
// MI455X (gfx1250) — matrix operand order repaired and checked by reading, not compiled, not run on hardware
//
#include <hip/hip_runtime.h>

// Problem: x [16, 512, 64, 64] f32, W [512, 512] f32
#define BATCH 16
#define CDIM  512
#define HWDIM 4096

typedef __attribute__((ext_vector_type(16))) __bf16 v16bf;
typedef __attribute__((ext_vector_type(8)))  __bf16 v8bf;
typedef __attribute__((ext_vector_type(8)))  float  v8f;
typedef __attribute__((ext_vector_type(2)))  float  v2f;
typedef __attribute__((ext_vector_type(4)))  int    v4i;

typedef __attribute__((address_space(1))) v4i as1_v4i;
typedef __attribute__((address_space(3))) v4i as3_v4i;

#if defined(__has_builtin)
#if __has_builtin(__builtin_amdgcn_global_load_async_to_lds_b128)
#define USE_ASYNC_LDS 1
#endif
#endif

// ---------- helpers ----------

__device__ __forceinline__ unsigned short f32_to_bf16_rne(float f) {
  union { float f; unsigned int u; } cv; cv.f = f;
  unsigned int u = cv.u;
  u += 0x7FFFu + ((u >> 16) & 1u);   // round to nearest even
  return (unsigned short)(u >> 16);
}

// A-operand fragment (16x16x32 bf16): lane half hv holds K = 8*hv..+7 (v0-3) and
// K = 16+8*hv..+7 (v4-7): lo chunk at p, hi chunk at p+16.
__device__ __forceinline__ v16bf frag_a(const unsigned short* p) {
  v8bf lo = *(const v8bf*)p;
  v8bf hi = *(const v8bf*)(p + 16);
  return __builtin_shufflevector(lo, hi, 0,1,2,3,4,5,6,7,8,9,10,11,12,13,14,15);
}
// B-operand fragment: lane half hv holds K = 16*hv..16*hv+15 contiguous.
__device__ __forceinline__ v16bf frag_b(const unsigned short* p) {
  v8bf lo = *(const v8bf*)p;
  v8bf hi = *(const v8bf*)(p + 8);
  return __builtin_shufflevector(lo, hi, 0,1,2,3,4,5,6,7,8,9,10,11,12,13,14,15);
}

__device__ __forceinline__ v8f wmma_bf16(v16bf a, v16bf b, v8f c) {
  return __builtin_amdgcn_wmma_f32_16x16x32_bf16(false, a, false, b, (short)0, c, false, false);
}

__device__ __forceinline__ float wave_max32(float v) {
#pragma unroll
  for (int m = 16; m > 0; m >>= 1) v = fmaxf(v, __shfl_xor(v, m, 32));
  return v;
}
__device__ __forceinline__ float wave_sum32(float v) {
#pragma unroll
  for (int m = 16; m > 0; m >>= 1) v += __shfl_xor(v, m, 32);
  return v;
}

// Stage a 128-row x 32-col bf16 panel (row-major, leading dim ld) into LDS.
// 512 x 16B chunks, 256 threads -> 2 async b128 copies per thread.
__device__ __forceinline__ void stage_tile(const unsigned short* gbase, size_t ld,
                                           unsigned short (*dst)[40], int tid) {
#pragma unroll
  for (int c = 0; c < 2; ++c) {
    const int chunk = tid + c * 256;
    const int row = chunk >> 2;
    const int col = (chunk & 3) * 8;
    const unsigned short* gp = gbase + (size_t)row * ld + col;
#if USE_ASYNC_LDS
    __builtin_amdgcn_global_load_async_to_lds_b128(
        (as1_v4i*)(v4i*)const_cast<unsigned short*>(gp),
        (as3_v4i*)(v4i*)&dst[row][col], 0, 0);
#else
    *(v8bf*)&dst[row][col] = *(const v8bf*)gp;
#endif
  }
}

__device__ __forceinline__ void wait_stage() {
#if USE_ASYNC_LDS
  asm volatile("s_wait_asynccnt 0x0" ::: "memory");
#endif
  __syncthreads();
}

// One K=32 step: wave (wm, wn) computes a 64x32 sub-tile = 4x2 WMMA tiles from LDS.
__device__ __forceinline__ void compute_step(const unsigned short (*At)[40],
                                             const unsigned short (*Bt)[40],
                                             int wm, int wn, int lm, int hv,
                                             v8f acc[4][2]) {
  const int aoff = hv * 8;
  const int boff = hv * 8;
  v16bf af[4], bf[2];
#pragma unroll
  for (int m = 0; m < 4; ++m) af[m] = frag_a(&At[wm * 64 + m * 16 + lm][aoff]);
#pragma unroll
  for (int n = 0; n < 2; ++n) bf[n] = frag_a(&Bt[wn * 32 + n * 16 + lm][boff]);
#pragma unroll
  for (int m = 0; m < 4; ++m)
#pragma unroll
    for (int n = 0; n < 2; ++n)
      acc[m][n] = wmma_bf16(af[m], bf[n], acc[m][n]);
}

// ---------- kernel 1: f32 -> bf16 convert + per-batch transpose ----------
__global__ void k_convert(const float* __restrict__ x,
                          unsigned short* __restrict__ xb,
                          unsigned short* __restrict__ xbt) {
  __shared__ unsigned short tile[32][33];
  const int b  = blockIdx.z;
  const int c0 = blockIdx.y * 32;
  const int h0 = blockIdx.x * 32;
  const int tx = threadIdx.x;
  const int ty = threadIdx.y;
  const float*    xB   = x   + (size_t)b * CDIM * HWDIM;
  unsigned short* xbB  = xb  + (size_t)b * CDIM * HWDIM;
  unsigned short* xbtB = xbt + (size_t)b * HWDIM * CDIM;
#pragma unroll
  for (int i = 0; i < 32; i += 8) {
    const int c = c0 + ty + i;
    const unsigned short bv = f32_to_bf16_rne(xB[(size_t)c * HWDIM + h0 + tx]);
    xbB[(size_t)c * HWDIM + h0 + tx] = bv;
    tile[ty + i][tx] = bv;
  }
  __syncthreads();
#pragma unroll
  for (int i = 0; i < 32; i += 8) {
    const int h = h0 + ty + i;
    xbtB[(size_t)h * CDIM + c0 + tx] = tile[tx][ty + i];
  }
}

// ---------- kernel 2: G[b] = Xb * Xb^T (workgroup-tiled 128x128, LDS double-buffered) ----
__global__ __launch_bounds__(256) void k_gram(const unsigned short* __restrict__ xb,
                                              float* __restrict__ g) {
  __shared__ unsigned short As[2][128][40];
  __shared__ unsigned short Bs[2][128][40];
  const int tid  = threadIdx.x;
  const int lane = tid & 31, wave = tid >> 5;
  const int wm = wave >> 2, wn = wave & 3;      // 2 x 4 wave grid
  const int lm = lane & 15, hv = lane >> 4;
  const int b = blockIdx.y, t = blockIdx.x;     // 4 x 4 tiles of 128
  const int c0 = (t >> 2) * 128;
  const int d0 = (t & 3) * 128;
  const unsigned short* X = xb + (size_t)b * CDIM * HWDIM;
  const unsigned short* Abase = X + (size_t)c0 * HWDIM;  // rows c0..c0+127
  const unsigned short* Bbase = X + (size_t)d0 * HWDIM;  // rows d0..d0+127
  v8f acc[4][2] = {};

  stage_tile(Abase, HWDIM, As[0], tid);
  stage_tile(Bbase, HWDIM, Bs[0], tid);
  wait_stage();
  for (int k0 = 0; k0 < HWDIM; k0 += 32) {
    const int cur = (k0 >> 5) & 1;
    if (k0 + 32 < HWDIM) {
      stage_tile(Abase + k0 + 32, HWDIM, As[cur ^ 1], tid);
      stage_tile(Bbase + k0 + 32, HWDIM, Bs[cur ^ 1], tid);
    }
    compute_step(As[cur], Bs[cur], wm, wn, lm, hv, acc);
    wait_stage();
  }

  float* G = g + (size_t)b * CDIM * CDIM;
#pragma unroll
  for (int m = 0; m < 4; ++m) {
    const int row0 = c0 + wm * 64 + m * 16 + hv * 8;
#pragma unroll
    for (int n = 0; n < 2; ++n) {
      const int col = d0 + wn * 32 + n * 16 + lm;
#pragma unroll
      for (int r = 0; r < 8; ++r)
        G[(size_t)(row0 + r) * CDIM + col] = acc[m][n][r];
    }
  }
}

// ---------- kernel 3: S[b] = G[b] * W^T  (full-precision f32 WMMA 16x16x4) ----------
__global__ void k_corr(const float* __restrict__ g,
                       const float* __restrict__ w,
                       float* __restrict__ s) {
  const int lane = threadIdx.x;
  const int b    = blockIdx.y;
  const int t    = blockIdx.x;
  const int c0   = (t >> 5) * 16;
  const int d0   = (t & 31) * 16;
  const int lm   = lane & 15;
  const int koff = (lane >> 4) * 2;
  const float* Arow = g + (size_t)b * CDIM * CDIM + (size_t)(c0 + lm) * CDIM;
  const float* Brow = w + (size_t)(d0 + lm) * CDIM;   // B[k,n] = W[d0+n, k]
  v8f acc = {};
#pragma unroll 4
  for (int k0 = 0; k0 < CDIM; k0 += 4) {
    const v2f av = *(const v2f*)(Arow + k0 + koff);
    const v2f bv = *(const v2f*)(Brow + k0 + koff);
    acc = __builtin_amdgcn_wmma_f32_16x16x4_f32(false, av, false, bv, (short)0, acc, false, false);
  }
  float* S = s + (size_t)b * CDIM * CDIM;
  const int mo = (lane >> 4) * 8;
#pragma unroll
  for (int r = 0; r < 8; ++r)
    S[(size_t)(c0 + mo + r) * CDIM + d0 + lm] = acc[r];
}

// ---------- kernel 4: softmax over axis=2 (rows), in place ----------
__global__ void k_softmax_rows(float* __restrict__ s) {
  float* p = s + (size_t)blockIdx.x * CDIM;
  const int lane = threadIdx.x;
  float v[16];
  float m = -3.402823466e38f;
#pragma unroll
  for (int i = 0; i < 16; ++i) { v[i] = p[lane + 32 * i]; m = fmaxf(m, v[i]); }
  m = wave_max32(m);
  float sum = 0.f;
#pragma unroll
  for (int i = 0; i < 16; ++i) { v[i] = __expf(v[i] - m); sum += v[i]; }
  sum = wave_sum32(sum);
  const float inv = __frcp_rn(sum);
#pragma unroll
  for (int i = 0; i < 16; ++i) p[lane + 32 * i] = v[i] * inv;
}

// ---------- kernel 5: softmax over axis=1 (cols), writes A as bf16 ----------
__global__ void k_softmax_cols(const float* __restrict__ s,
                               unsigned short* __restrict__ ab) {
  const int col  = blockIdx.x;
  const int b    = col >> 9;
  const int d    = col & (CDIM - 1);
  const int lane = threadIdx.x;
  const float*    p = s  + (size_t)b * CDIM * CDIM + d;
  unsigned short* q = ab + (size_t)b * CDIM * CDIM + d;
  float v[16];
  float m = -3.402823466e38f;
#pragma unroll
  for (int i = 0; i < 16; ++i) { v[i] = p[(size_t)(lane + 32 * i) * CDIM]; m = fmaxf(m, v[i]); }
  m = wave_max32(m);
  float sum = 0.f;
#pragma unroll
  for (int i = 0; i < 16; ++i) { v[i] = __expf(v[i] - m); sum += v[i]; }
  sum = wave_sum32(sum);
  const float inv = __frcp_rn(sum);
#pragma unroll
  for (int i = 0; i < 16; ++i)
    q[(size_t)(lane + 32 * i) * CDIM] = f32_to_bf16_rne(v[i] * inv);
}

// ---------- kernel 6: out[b] = A[b] * X[b] + X[b] (workgroup-tiled, LDS staged) ----------
__global__ __launch_bounds__(256) void k_out(const unsigned short* __restrict__ ab,
                                             const unsigned short* __restrict__ xbt,
                                             const float* __restrict__ x,
                                             float* __restrict__ out) {
  __shared__ unsigned short As[2][128][40];
  __shared__ unsigned short Bs[2][128][40];
  const int tid  = threadIdx.x;
  const int lane = tid & 31, wave = tid >> 5;
  const int wm = wave >> 2, wn = wave & 3;
  const int lm = lane & 15, hv = lane >> 4;
  const int b = blockIdx.y, t = blockIdx.x;     // (512/128) x (4096/128) = 4 x 32
  const int c0 = (t >> 5) * 128;
  const int h0 = (t & 31) * 128;
  const unsigned short* Abase = ab  + (size_t)b * CDIM * CDIM  + (size_t)c0 * CDIM;
  const unsigned short* Bbase = xbt + (size_t)b * HWDIM * CDIM + (size_t)h0 * CDIM;
  v8f acc[4][2] = {};

  stage_tile(Abase, CDIM, As[0], tid);
  stage_tile(Bbase, CDIM, Bs[0], tid);
  wait_stage();
  for (int k0 = 0; k0 < CDIM; k0 += 32) {
    const int cur = (k0 >> 5) & 1;
    if (k0 + 32 < CDIM) {
      stage_tile(Abase + k0 + 32, CDIM, As[cur ^ 1], tid);
      stage_tile(Bbase + k0 + 32, CDIM, Bs[cur ^ 1], tid);
    }
    compute_step(As[cur], Bs[cur], wm, wn, lm, hv, acc);
    wait_stage();
  }

  const float* xB = x   + (size_t)b * CDIM * HWDIM;
  float*       oB = out + (size_t)b * CDIM * HWDIM;
#pragma unroll
  for (int m = 0; m < 4; ++m) {
    const int row0 = c0 + wm * 64 + m * 16 + hv * 8;
#pragma unroll
    for (int n = 0; n < 2; ++n) {
      const int col = h0 + wn * 32 + n * 16 + lm;
#pragma unroll
      for (int r = 0; r < 8; ++r) {
        const size_t idx = (size_t)(row0 + r) * HWDIM + col;
        oB[idx] = acc[m][n][r] + xB[idx];
      }
    }
  }
}

// ---------- host launcher ----------
extern "C" void kernel_launch(void* const* d_in, const int* in_sizes, int n_in,
                              void* d_out, int out_size, void* d_ws, size_t ws_size,
                              hipStream_t stream) {
  const float* x  = (const float*)d_in[0];
  const float* Wm = (const float*)d_in[1];
  float* out = (float*)d_out;

  char* ws = (char*)d_ws;
  const size_t SZ_XB = (size_t)BATCH * CDIM * HWDIM * sizeof(unsigned short); // 64 MiB
  const size_t SZ_G  = (size_t)BATCH * CDIM * CDIM * sizeof(float);           // 16 MiB
  unsigned short* xb  = (unsigned short*)(ws);
  unsigned short* xbt = (unsigned short*)(ws + SZ_XB);
  float*          G   = (float*)(ws + 2 * SZ_XB);
  float*          S   = (float*)(ws + 2 * SZ_XB + SZ_G);
  unsigned short* Ab  = (unsigned short*)(ws + 2 * SZ_XB + 2 * SZ_G);

  k_convert<<<dim3(HWDIM / 32, CDIM / 32, BATCH), dim3(32, 8), 0, stream>>>(x, xb, xbt);
  k_gram<<<dim3(16, BATCH), dim3(256), 0, stream>>>(xb, G);
  k_corr<<<dim3(1024, BATCH), dim3(32), 0, stream>>>(G, Wm, S);
  k_softmax_rows<<<dim3(BATCH * CDIM), dim3(32), 0, stream>>>(S);
  k_softmax_cols<<<dim3(BATCH * CDIM), dim3(32), 0, stream>>>(S, Ab);
  k_out<<<dim3(128, BATCH), dim3(256), 0, stream>>>(Ab, xbt, x, out);
}
